// RNNModel_52140902974330
// MI455X (gfx1250) — matrix operand order repaired and checked by reading, not compiled, not run on hardware
//
#include <hip/hip_runtime.h>
#include <hip/hip_bf16.h>
#include <math.h>

typedef _Float16 half_t;
typedef __attribute__((ext_vector_type(16))) _Float16 v16h;
typedef __attribute__((ext_vector_type(8)))  _Float16 v8h;
typedef __attribute__((ext_vector_type(8)))  float    v8f;

#define LSTM_B   32
#define LSTM_T   512
#define LSTM_IN  512
#define LSTM_H   512
#define LSTM_G   (4*LSTM_H)      // 2048 gate columns
#define LSTM_M   (LSTM_T*LSTM_B) // 16384 rows for the big GEMMs

// ---------------------------------------------------------------- WMMA core
__device__ __forceinline__ v8f wmma_f16(v16h a, v16h b, v8f c) {
  // D = A(16x32 f16) * B(32x16 f16) + C(16x16 f32)
  return __builtin_amdgcn_wmma_f32_16x16x32_f16(false, a, false, b, (short)0, c,
                                                false, false);
}

// A-matrix 16x32 f16 fragment from a row-major [*, ld] f16 matrix (global or LDS).
__device__ __forceinline__ v16h load_a16x32(const half_t* base, int ld,
                                            int m0, int k0, int lane) {
  int row = m0 + (lane & 15);
  int kb  = k0 + ((lane >> 4) << 4);
  const half_t* p = base + (size_t)row * ld + kb;
  v8h lo = *(const v8h*)(p);
  v8h hi = *(const v8h*)(p + 8);
  v16h r;
#pragma unroll
  for (int i = 0; i < 8; ++i) { r[i] = lo[i]; r[i + 8] = hi[i]; }
  return r;
}

// B-matrix 32x16 f16 fragment where B[k][n] = W[n0+n][k0+k], W row-major [N,ld].
__device__ __forceinline__ v16h load_b32x16(const half_t* w, int ld,
                                            int n0, int k0, int lane) {
  int col = n0 + (lane & 15);
  int kb  = k0 + ((lane >> 4) << 4);
  return *(const v16h*)(w + (size_t)col * ld + kb);
}

// ------------------------------------------- CDNA5 async global->LDS staging
#if __has_builtin(__builtin_amdgcn_global_load_async_to_lds_b128) || \
    __has_builtin(__builtin_amdgcn_load_to_lds)
#define LSTM_ASYNC_COPY 1
#endif

typedef int b128_t __attribute__((vector_size(16)));
typedef __attribute__((address_space(1))) b128_t* g_b128_p;
typedef __attribute__((address_space(3))) b128_t* l_b128_p;

__device__ __forceinline__ void cp_g2l_b128(const half_t* g, half_t* l) {
#if __has_builtin(__builtin_amdgcn_global_load_async_to_lds_b128)
  __builtin_amdgcn_global_load_async_to_lds_b128((g_b128_p)(void*)(size_t)(const void*)g,
                                                 (l_b128_p)l, 0, 0);
#elif __has_builtin(__builtin_amdgcn_load_to_lds)
  typedef __attribute__((address_space(3))) void lv_t;
  __builtin_amdgcn_load_to_lds((void*)g, (lv_t*)l, 16, 0, 0);
#else
  *(v8h*)l = *(const v8h*)g;
#endif
}

__device__ __forceinline__ void async_wait_all() {
#ifdef LSTM_ASYNC_COPY
#if __has_builtin(__builtin_amdgcn_s_wait_asynccnt)
  __builtin_amdgcn_s_wait_asynccnt(0);
#else
  asm volatile("s_wait_asynccnt 0" ::: "memory");
#endif
#endif
}

// -------------------------------------------------------- conversion kernels
__global__ void k_cvt_f32_to_f16(const float* __restrict__ in,
                                 half_t* __restrict__ out, size_t n) {
  for (size_t i = (size_t)blockIdx.x * blockDim.x + threadIdx.x; i < n;
       i += (size_t)gridDim.x * blockDim.x)
    out[i] = (half_t)in[i];
}

// x[b][t][d] (f32) -> xT[(t*B+b)][d] (f16)
__global__ void k_transpose_x(const float* __restrict__ x,
                              half_t* __restrict__ xT) {
  size_t total = (size_t)LSTM_B * LSTM_T * LSTM_IN;
  for (size_t i = (size_t)blockIdx.x * blockDim.x + threadIdx.x; i < total;
       i += (size_t)gridDim.x * blockDim.x) {
    int d = (int)(i % LSTM_IN);
    size_t bt = i / LSTM_IN;
    int t = (int)(bt % LSTM_T);
    int b = (int)(bt / LSTM_T);
    xT[((size_t)t * LSTM_B + b) * LSTM_IN + d] = (half_t)x[i];
  }
}

__global__ void k_init_state(half_t* __restrict__ h16,
                             float* __restrict__ h32,
                             float* __restrict__ cbuf,
                             unsigned* __restrict__ bar) {
  int i = blockIdx.x * blockDim.x + threadIdx.x;
  if (i < 2 * 2 * LSTM_B * LSTM_H) h16[i] = (half_t)0.f;   // both parity buffers
  if (i < 2 * LSTM_B * LSTM_H) { h32[i] = 0.f; cbuf[i] = 0.f; }
  if (i < 4) bar[i] = 0u;
}

__global__ void k_copy_states(const float* __restrict__ h32,
                              const float* __restrict__ cbuf,
                              float* __restrict__ hn, float* __restrict__ cn) {
  int i = blockIdx.x * blockDim.x + threadIdx.x;
  if (i < 2 * LSTM_B * LSTM_H) { hn[i] = h32[i]; cn[i] = cbuf[i]; }
}

// ------------------------------------------------- big GEMM: xg = A*W^T + b
// A: [M,K] f16 row-major.  W: per-dir [N,K] f16 row-major.  xg: per-dir [M,N] f16.
// Block = 256 thr (8 waves, 2x4), block tile 64(M) x 256(N), wave tile 32x64.
__global__ void k_gemm_xg(const half_t* __restrict__ A,
                          const half_t* __restrict__ W,
                          const float* __restrict__ bx,
                          const float* __restrict__ bh,
                          half_t* __restrict__ xg,
                          int M, int N, int K) {
  int dir = blockIdx.z;
  const half_t* Wd  = W  + (size_t)dir * N * K;
  const float*  bxd = bx + (size_t)dir * N;
  const float*  bhd = bh + (size_t)dir * N;
  half_t*       xgd = xg + (size_t)dir * M * N;

  int lane = threadIdx.x & 31;
  int wave = threadIdx.x >> 5;
  int m0 = blockIdx.x * 64 + (wave & 1) * 32;
  int n0 = blockIdx.y * 256 + (wave >> 1) * 64;

  v8f acc[2][4] = {};
  for (int k = 0; k < K; k += 32) {
    v16h a0 = load_a16x32(A, K, m0,      k, lane);
    v16h a1 = load_a16x32(A, K, m0 + 16, k, lane);
    if (k + 32 < K)  // near-cache prefetch of next A chunk
      __builtin_prefetch(A + (size_t)(m0 + (lane & 15)) * K + k + 32, 0, 3);
#pragma unroll
    for (int j = 0; j < 4; ++j) {
      v16h b = load_b32x16(Wd, K, n0 + j * 16, k, lane);
      acc[0][j] = wmma_f16(a0, b, acc[0][j]);
      acc[1][j] = wmma_f16(a1, b, acc[1][j]);
    }
  }
#pragma unroll
  for (int j = 0; j < 4; ++j) {
    int col = n0 + j * 16 + (lane & 15);
    float bias = bxd[col] + bhd[col];
#pragma unroll
    for (int mt = 0; mt < 2; ++mt) {
      int rowbase = m0 + mt * 16 + ((lane >> 4) << 3);
#pragma unroll
      for (int r = 0; r < 8; ++r)
        xgd[(size_t)(rowbase + r) * N + col] = (half_t)(acc[mt][j][r] + bias);
    }
  }
}

// ------------------------------------- persistent LSTM layer (all timesteps)
// grid (H/32, 2 dirs), block 256 (8 waves). Block owns h-columns [c0,c0+32) of
// one dir. Per step: async-stage h_prev to LDS, WMMA the 4x32 gate columns
// over 32 batch rows, fuse gating, write h, then inter-block barrier (per dir,
// monotonic agent-scope atomic counter: 16 blocks/dir, all co-resident).
template <bool LAST>
__global__ void k_lstm_layer(const half_t* __restrict__ xg,   // [dir][T*B][4H] f16
                             const half_t* __restrict__ Wh,   // [dir][4H][H]  f16
                             half_t* __restrict__ h16,        // [dir][2parity][B][H]
                             float*  __restrict__ h32,        // [dir][B][H] (final)
                             float*  __restrict__ cbuf,       // [dir][B][H]
                             half_t* __restrict__ outseq,     // [T*B][2H] f16 (layer0)
                             float*  __restrict__ outfin,     // [B][T][2H] f32 (layer1)
                             unsigned* __restrict__ bar) {    // [2] per-dir counters
  __shared__ half_t sh[LSTM_B * LSTM_H];       // 32 KB: h_prev staged in LDS
  __shared__ float  sg[4][LSTM_B][32];         // 16 KB: raw gate pre-activations

  const int dir = blockIdx.y;
  const int c0  = blockIdx.x * 32;
  const int lane = threadIdx.x & 31;
  const int wave = threadIdx.x >> 5;
  const int gate = wave >> 1;                  // 0..3 : i,f,g,o
  const int nloc = (wave & 1) * 16;            // 0 or 16 within the 32-col group
  const int gcol0 = gate * LSTM_H + c0 + nloc; // column in 4H gate space
  const half_t* Whd = Wh + (size_t)dir * LSTM_G * LSTM_H;
  unsigned* mybar = bar + dir;

  for (int step = 0; step < LSTM_T; ++step) {
    const int t = (dir == 0) ? step : (LSTM_T - 1 - step);
    const half_t* hp = h16 + ((size_t)dir * 2 + (step & 1))       * LSTM_B * LSTM_H;
    half_t*       hn = h16 + ((size_t)dir * 2 + ((step + 1) & 1)) * LSTM_B * LSTM_H;

    // stage h_prev (32x512 f16 = 32KB) into LDS via async global->LDS
    for (int i = threadIdx.x; i < LSTM_B * LSTM_H / 8; i += 256)
      cp_g2l_b128(hp + (size_t)i * 8, sh + (size_t)i * 8);
    async_wait_all();
    __syncthreads();

    v8f acc[2] = {};
    for (int k = 0; k < LSTM_H; k += 32) {
      v16h b = load_b32x16(Whd, LSTM_H, gcol0, k, lane);  // L2-resident weights
      v16h a0 = load_a16x32(sh, LSTM_H, 0,  k, lane);     // LDS reads
      v16h a1 = load_a16x32(sh, LSTM_H, 16, k, lane);
      acc[0] = wmma_f16(a0, b, acc[0]);
      acc[1] = wmma_f16(a1, b, acc[1]);
    }

    // scatter accumulators to LDS gate buffer
#pragma unroll
    for (int mt = 0; mt < 2; ++mt) {
      int rowbase = mt * 16 + ((lane >> 4) << 3);
      int colL = nloc + (lane & 15);
#pragma unroll
      for (int r = 0; r < 8; ++r) sg[gate][rowbase + r][colL] = acc[mt][r];
    }
    __syncthreads();

    // fused gating: 32 rows x 32 cols, 4 elems/thread
    for (int e = threadIdx.x; e < LSTM_B * 32; e += 256) {
      int row = e >> 5, col = e & 31;
      const half_t* xgr = xg + ((size_t)dir * LSTM_T * LSTM_B +
                                (size_t)t * LSTM_B + row) * LSTM_G + c0 + col;
      float gi = sg[0][row][col] + (float)xgr[0 * LSTM_H];
      float gf = sg[1][row][col] + (float)xgr[1 * LSTM_H];
      float gg = sg[2][row][col] + (float)xgr[2 * LSTM_H];
      float go = sg[3][row][col] + (float)xgr[3 * LSTM_H];
      float ii = 1.f / (1.f + __expf(-gi));
      float ff = 1.f / (1.f + __expf(-gf));
      float tg = tanhf(gg);
      float oo = 1.f / (1.f + __expf(-go));
      float* cp = cbuf + ((size_t)dir * LSTM_B + row) * LSTM_H + c0 + col;
      float cn = ff * (*cp) + ii * tg;   // c columns are block-exclusive: in place
      float hv = oo * tanhf(cn);
      *cp = cn;
      hn[(size_t)row * LSTM_H + c0 + col] = (half_t)hv;
      if (!LAST) {
        outseq[((size_t)t * LSTM_B + row) * (2 * LSTM_H) + dir * LSTM_H + c0 + col] =
            (half_t)hv;
      } else {
        outfin[((size_t)row * LSTM_T + t) * (2 * LSTM_H) + dir * LSTM_H + c0 + col] = hv;
      }
      if (step == LSTM_T - 1)
        h32[((size_t)dir * LSTM_B + row) * LSTM_H + c0 + col] = hv;
    }

    // ---- inter-block barrier for this direction (16 blocks, monotonic count)
    __threadfence();                   // release this block's h writes (agent)
    __syncthreads();
    if (threadIdx.x == 0) {
      __hip_atomic_fetch_add(mybar, 1u, __ATOMIC_RELEASE, __HIP_MEMORY_SCOPE_AGENT);
      unsigned target = 16u * (unsigned)(step + 1);
      while (__hip_atomic_load(mybar, __ATOMIC_ACQUIRE, __HIP_MEMORY_SCOPE_AGENT) <
             target)
        __builtin_amdgcn_s_sleep(2);
    }
    __syncthreads();
    __threadfence();                   // acquire other blocks' h writes
  }
}

// --------------------------------------------------------------------- host
extern "C" void kernel_launch(void* const* d_in, const int* in_sizes, int n_in,
                              void* d_out, int out_size, void* d_ws, size_t ws_size,
                              hipStream_t stream) {
  const float* x   = (const float*)d_in[0];
  const float* Wx0 = (const float*)d_in[1];
  const float* Wh0 = (const float*)d_in[2];
  const float* bx0 = (const float*)d_in[3];
  const float* bh0 = (const float*)d_in[4];
  const float* Wx1 = (const float*)d_in[5];
  const float* Wh1 = (const float*)d_in[6];
  const float* bx1 = (const float*)d_in[7];
  const float* bh1 = (const float*)d_in[8];
  float* out = (float*)d_out;

  char* ws = (char*)d_ws;
  size_t off = 0;
  auto carve = [&](size_t bytes) -> char* {
    char* p = ws + off;
    off += (bytes + 255) & ~(size_t)255;
    return p;
  };
  half_t* x16   = (half_t*)carve((size_t)LSTM_M * LSTM_IN * 2);            // 16 MB
  half_t* wx0h  = (half_t*)carve((size_t)2 * LSTM_G * LSTM_IN * 2);        //  4 MB
  half_t* wh0h  = (half_t*)carve((size_t)2 * LSTM_G * LSTM_H * 2);         //  4 MB
  half_t* wx1h  = (half_t*)carve((size_t)2 * LSTM_G * (2 * LSTM_H) * 2);   //  8 MB
  half_t* wh1h  = (half_t*)carve((size_t)2 * LSTM_G * LSTM_H * 2);         //  4 MB
  half_t* xgbuf = (half_t*)carve((size_t)2 * LSTM_M * LSTM_G * 2);         // 134 MB
  half_t* out0  = (half_t*)carve((size_t)LSTM_M * 2 * LSTM_H * 2);         // 32 MB
  half_t* h16   = (half_t*)carve((size_t)2 * 2 * LSTM_B * LSTM_H * 2);
  float*  h32   = (float*)carve((size_t)2 * LSTM_B * LSTM_H * 4);
  float*  cbuf  = (float*)carve((size_t)2 * LSTM_B * LSTM_H * 4);
  unsigned* bar = (unsigned*)carve(4 * sizeof(unsigned));
  (void)ws_size; (void)in_sizes; (void)n_in; (void)out_size;

  const size_t OUT_ELEMS = (size_t)LSTM_B * LSTM_T * 2 * LSTM_H; // 16,777,216
  float* hn_out = out + OUT_ELEMS;
  float* cn_out = hn_out + (size_t)4 * LSTM_B * LSTM_H;

  // ---- conversions
  k_transpose_x<<<2048, 256, 0, stream>>>(x, x16);
  k_cvt_f32_to_f16<<<1024, 256, 0, stream>>>(Wx0, wx0h, (size_t)2 * LSTM_G * LSTM_IN);
  k_cvt_f32_to_f16<<<1024, 256, 0, stream>>>(Wh0, wh0h, (size_t)2 * LSTM_G * LSTM_H);
  k_cvt_f32_to_f16<<<2048, 256, 0, stream>>>(Wx1, wx1h, (size_t)2 * LSTM_G * 2 * LSTM_H);
  k_cvt_f32_to_f16<<<1024, 256, 0, stream>>>(Wh1, wh1h, (size_t)2 * LSTM_G * LSTM_H);

  // ---- layer 0
  k_init_state<<<(2 * 2 * LSTM_B * LSTM_H + 255) / 256, 256, 0, stream>>>(h16, h32,
                                                                          cbuf, bar);
  k_gemm_xg<<<dim3(LSTM_M / 64, LSTM_G / 256, 2), 256, 0, stream>>>(
      x16, wx0h, bx0, bh0, xgbuf, LSTM_M, LSTM_G, LSTM_IN);
  k_lstm_layer<false><<<dim3(LSTM_H / 32, 2), 256, 0, stream>>>(
      xgbuf, wh0h, h16, h32, cbuf, out0, out, bar);
  k_copy_states<<<(2 * LSTM_B * LSTM_H + 255) / 256, 256, 0, stream>>>(
      h32, cbuf, hn_out, cn_out);

  // ---- layer 1
  k_init_state<<<(2 * 2 * LSTM_B * LSTM_H + 255) / 256, 256, 0, stream>>>(h16, h32,
                                                                          cbuf, bar + 2);
  k_gemm_xg<<<dim3(LSTM_M / 64, LSTM_G / 256, 2), 256, 0, stream>>>(
      out0, wx1h, bx1, bh1, xgbuf, LSTM_M, LSTM_G, 2 * LSTM_H);
  k_lstm_layer<true><<<dim3(LSTM_H / 32, 2), 256, 0, stream>>>(
      xgbuf, wh1h, h16, h32, cbuf, out0, out, bar + 2);
  k_copy_states<<<(2 * LSTM_B * LSTM_H + 255) / 256, 256, 0, stream>>>(
      h32, cbuf, hn_out + 2 * LSTM_B * LSTM_H, cn_out + 2 * LSTM_B * LSTM_H);
}
